// VectorQuantizer_66571993088827
// MI455X (gfx1250) — matrix operand order repaired and checked by reading, not compiled, not run on hardware
//
#include <hip/hip_runtime.h>
#include <hip/hip_bf16.h>

// ---------------------------------------------------------------------------
// VQ-VAE vector quantizer for MI455X (gfx1250, wave32, WMMA + TDM).
//
// z (16,128,64,64) f32, codebook (1024,128) f32.
// Outputs (float, concatenated):
//   [0..8388607] quantized | [8388608..8454143] indices | [8454144] vq_loss
//   [8454145] perplexity
// ---------------------------------------------------------------------------

#define C_DIM    128
#define K_CODES  1024
#define N_PTS    65536
#define Z_ELEMS  8388608
#define CHUNK    32              // codes per TDM chunk
#define NCHUNK   (K_CODES / CHUNK)
#define CSTR     136             // LDS row stride in bf16 elems (TDM pad: 256B+16B)
#define BUFELEMS (2 * CHUNK * CSTR)   // hi+lo planes of one chunk (8704 ushort)
#define GATHER_BLOCKS 32768      // Z_ELEMS / 256

typedef __attribute__((ext_vector_type(16))) __bf16 bf16x16;
typedef __attribute__((ext_vector_type(8)))  float  floatx8;
typedef __attribute__((ext_vector_type(4)))  unsigned int v4u;
typedef __attribute__((ext_vector_type(8)))  int          v8i;
typedef __attribute__((ext_vector_type(4)))  int          v4i;

union WmmaOp { bf16x16 v; uint4 q[2]; };

// round-to-nearest-even f32 -> bf16 bits
__device__ __forceinline__ unsigned int f2bf_bits(float f) {
    unsigned int u = __float_as_uint(f);
    return (u + 0x7FFFu + ((u >> 16) & 1u)) >> 16;
}

// -------------------------------- init -------------------------------------
__global__ __launch_bounds__(256) void vq_init_kernel(unsigned int* __restrict__ counts) {
    int t = blockIdx.x * 256 + threadIdx.x;
    if (t < K_CODES) counts[t] = 0u;
}

// ------------------------------- ||c||^2 -----------------------------------
__global__ __launch_bounds__(256) void vq_cnorm_kernel(const float* __restrict__ cb,
                                                       float* __restrict__ cnorm) {
    int k = blockIdx.x * 256 + threadIdx.x;
    if (k < K_CODES) {
        const float* r = cb + (size_t)k * C_DIM;
        float s = 0.f;
        #pragma unroll 8
        for (int c = 0; c < C_DIM; ++c) s = fmaf(r[c], r[c], s);
        cnorm[k] = s;
    }
}

// ---------------- codebook f32 -> bf16 hi/lo planes (TDM layout) ------------
// Layout: [chunk][plane 0=hi,1=lo][code-in-chunk 32][ch 128] ushort.
__global__ __launch_bounds__(256) void vq_cbsplit_kernel(const float* __restrict__ cb,
                                                         unsigned short* __restrict__ cbs) {
    int t = blockIdx.x * 256 + threadIdx.x;       // 512 blocks -> 131072 elems
    int k = t >> 7, c = t & 127;
    float f = cb[t];
    unsigned int hb = f2bf_bits(f);
    float hf = __uint_as_float(hb << 16);
    unsigned int lb = f2bf_bits(f - hf);
    int chunk = k >> 5, i = k & 31;
    size_t base = ((size_t)chunk * 2 * CHUNK) << 7;
    cbs[base + (((size_t)0 * CHUNK + i) << 7) + c] = (unsigned short)hb;
    cbs[base + (((size_t)1 * CHUNK + i) << 7) + c] = (unsigned short)lb;
}

// --------------------- TDM issue: one codebook chunk -> LDS -----------------
// 3D tile: x = 128 ch (2B), y = 32 codes (stride 128), z = 2 planes
// (stride 4096).  pad_enable: +4 DWORDs after every 64 DWORDs => LDS row
// stride 272B = CSTR bf16 elems.
__device__ __forceinline__ void tdm_load_chunk(const unsigned short* gsrc,
                                               unsigned int lds_byte_off) {
    unsigned long long ga = (unsigned long long)(const void*)gsrc;
    v4u g0;
    g0.x = 1u;                                    // count=1, user descriptor
    g0.y = lds_byte_off;                          // D#.lds_addr
    g0.z = (unsigned int)ga;                      // D#.global_addr[31:0]
    g0.w = (unsigned int)(ga >> 32) | (2u << 30); // addr[56:32] | type=2
    v8i g1;
    g1[0] = (1 << 16)       // data_size = 2 bytes
          | (1 << 20)       // pad_enable
          | (5 << 22)       // pad_interval: 64 DWORDs (one 256B row)
          | (3 << 25);      // pad_amount: 4 DWORDs (16B)
    g1[1] = (int)(128u << 16);        // tensor_dim0 = 128 (low 16 in [63:48])
    g1[2] = (int)(32u << 16);         // dim0 hi16=0 | tensor_dim1 = 32
    g1[3] = (int)(128u << 16);        // dim1 hi16=0 | tile_dim0 = 128
    g1[4] = (int)(32u | (2u << 16));  // tile_dim1 = 32 | tile_dim2 = 2
    g1[5] = 128;                      // tensor_dim0_stride low32
    g1[6] = (int)(4096u << 16);       // d0s hi16=0 | tensor_dim1_stride low16
    g1[7] = 0;                        // tensor_dim1_stride high32
    v4i g2 = {2, 0, 0, 0};            // tensor_dim2 = 2 (3D)
    v4i g3 = {0, 0, 0, 0};
    v8i g4 = {0, 0, 0, 0, 0, 0, 0, 0};   // unused trailing SGPR group (clang-23 form)
    __builtin_amdgcn_tensor_load_to_lds(g0, g1, g2, g3, g4, 0);
}

// ------------------------- distance GEMM + argmin ---------------------------
// 512 blocks x 256 threads (8 waves).  Block g owns rows [g*128, g*128+128).
// argmin_k(||c_k||^2 - 2 z.c_k)  (||z||^2 is row-constant, dropped).
union ArgminSmem {
    unsigned short zplane[2][128 * C_DIM];   // phase 1: z tile hi/lo (64 KB)
    unsigned short cbbuf[2][BUFELEMS];       // phase 2: double-buffered chunks
};

__global__ __launch_bounds__(256) void vq_argmin_kernel(
    const float* __restrict__ z, const unsigned short* __restrict__ cbs,
    const float* __restrict__ cnorm, float* __restrict__ idx_out,
    unsigned int* __restrict__ counts)
{
    __shared__ ArgminSmem sm;

    const int tid  = threadIdx.x;
    const int lane = tid & 31;
    const int wave = tid >> 5;
    const int g    = blockIdx.x;
    const int b    = g >> 5;
    const int hw0  = (g & 31) << 7;

    // ---- phase 1: transpose-load z tile, split f32 -> bf16 hi + lo ----
    const float* zb = z + ((size_t)b << 19);
    for (int t = tid; t < 128 * C_DIM; t += 256) {
        int c = t >> 7, r = t & 127;
        float f  = zb[((size_t)c << 12) + (size_t)(hw0 + r)];
        unsigned int hb = f2bf_bits(f);
        float hf = __uint_as_float(hb << 16);
        unsigned int lb = f2bf_bits(f - hf);
        sm.zplane[0][r * C_DIM + c] = (unsigned short)hb;
        sm.zplane[1][r * C_DIM + c] = (unsigned short)lb;
    }
    __syncthreads();

    // ---- build A operands (16x32 bf16, ISA 7.12.2 layout) in registers ----
    WmmaOp ahi[4], alo[4];
    {
        int row = (wave << 4) + (lane & 15);
        int hi8 = (lane >> 4) << 3;
        #pragma unroll
        for (int ks = 0; ks < 4; ++ks) {
            int base = row * C_DIM + (ks << 5) + hi8;
            ahi[ks].q[0] = *(const uint4*)&sm.zplane[0][base];
            ahi[ks].q[1] = *(const uint4*)&sm.zplane[0][base + 16];
            alo[ks].q[0] = *(const uint4*)&sm.zplane[1][base];
            alo[ks].q[1] = *(const uint4*)&sm.zplane[1][base + 16];
        }
    }
    __syncthreads();   // everyone done reading z planes; LDS may be reused

    const unsigned int ldsbuf0 = (unsigned int)(unsigned long long)(const void*)&sm.cbbuf[0][0];
    const unsigned int ldsbuf1 = (unsigned int)(unsigned long long)(const void*)&sm.cbbuf[1][0];

    // prologue: wave 0 streams chunk 0 into buffer 0 via the TDM
    if (wave == 0) tdm_load_chunk(cbs, ldsbuf0);

    float minv[8];
    int   mini[8];
    #pragma unroll
    for (int j = 0; j < 8; ++j) { minv[j] = 3.4e38f; mini[j] = 0; }

    for (int chunk = 0; chunk < NCHUNK; ++chunk) {
        const int cur = chunk & 1;
        if (wave == 0) __builtin_amdgcn_s_wait_tensorcnt(0);  // chunk data landed
        __syncthreads();   // data visible to all waves; prev buffer fully consumed
        if (wave == 0 && chunk + 1 < NCHUNK)
            tdm_load_chunk(cbs + (size_t)(chunk + 1) * (2 * CHUNK * C_DIM),
                           cur ? ldsbuf0 : ldsbuf1);

        const unsigned short* buf = sm.cbbuf[cur];
        #pragma unroll
        for (int nt = 0; nt < CHUNK / 16; ++nt) {
            floatx8 acc = {0.f, 0.f, 0.f, 0.f, 0.f, 0.f, 0.f, 0.f};
            #pragma unroll
            for (int ks = 0; ks < 4; ++ks) {
                // B operand (32x16 bf16): lane L slot e: K = kb+8*(L>=16)+e (e<8),
                // kb+16+8*(L>=16)+(e-8) (e>=8), N = L%16
                int off = ((nt << 4) + (lane & 15)) * CSTR + (ks << 5) + ((lane >> 4) << 3);
                WmmaOp bh, bl;
                bh.q[0] = *(const uint4*)&buf[off];
                bh.q[1] = *(const uint4*)&buf[off + 16];
                bl.q[0] = *(const uint4*)&buf[off + CHUNK * CSTR];
                bl.q[1] = *(const uint4*)&buf[off + CHUNK * CSTR + 16];
                // 3-term split product: ~f32 accuracy on the bf16 pipe
                acc = __builtin_amdgcn_wmma_f32_16x16x32_bf16(
                          false, ahi[ks].v, false, bh.v, (short)0, acc, false, false);
                acc = __builtin_amdgcn_wmma_f32_16x16x32_bf16(
                          false, alo[ks].v, false, bh.v, (short)0, acc, false, false);
                acc = __builtin_amdgcn_wmma_f32_16x16x32_bf16(
                          false, ahi[ks].v, false, bl.v, (short)0, acc, false, false);
            }
            int   code = (chunk << 5) + (nt << 4) + (lane & 15);
            float cn   = cnorm[code];
            #pragma unroll
            for (int j = 0; j < 8; ++j) {          // C/D layout: M = j + 8*(lane>=16)
                float d = fmaf(-2.f, acc[j], cn);
                if (d < minv[j]) { minv[j] = d; mini[j] = code; }
            }
        }
    }

    // ---- min+index reduction across the 16 lanes holding the same M rows ----
    #pragma unroll
    for (int m = 1; m <= 8; m <<= 1) {
        #pragma unroll
        for (int j = 0; j < 8; ++j) {
            float ov = __shfl_xor(minv[j], m, 32);
            int   oi = __shfl_xor(mini[j], m, 32);
            if (ov < minv[j] || (ov == minv[j] && oi < mini[j])) {
                minv[j] = ov; mini[j] = oi;
            }
        }
    }
    if ((lane & 15) == 0) {
        int rowbase = (g << 7) + (wave << 4) + ((lane >> 4) << 3);
        #pragma unroll
        for (int j = 0; j < 8; ++j) {
            idx_out[rowbase + j] = (float)mini[j];
            atomicAdd(&counts[mini[j]], 1u);       // integer atomics: deterministic
        }
    }
}

// --------------------- gather + straight-through + MSE ----------------------
__global__ __launch_bounds__(256) void vq_gather_kernel(
    const float* __restrict__ z, const float* __restrict__ codebook,
    const float* __restrict__ idxF, float* __restrict__ qout,
    float* __restrict__ partials)
{
    __shared__ float wsum[8];
    size_t e = (size_t)blockIdx.x * 256 + threadIdx.x;
    int hw = (int)(e & 4095);
    int c  = (int)((e >> 12) & 127);
    int b  = (int)(e >> 19);
    int n  = (b << 12) + hw;
    int code = (int)idxF[n];
    float q  = codebook[(size_t)code * C_DIM + c];
    float zv = z[e];
    float qd = q - zv;
    qout[e] = zv + qd;                  // z + (q_raw - z): straight-through value
    float d2 = qd * qd;
    #pragma unroll
    for (int m = 16; m > 0; m >>= 1) d2 += __shfl_down(d2, m, 32);
    if ((threadIdx.x & 31) == 0) wsum[threadIdx.x >> 5] = d2;
    __syncthreads();
    if (threadIdx.x < 32) {
        float v = (threadIdx.x < 8) ? wsum[threadIdx.x] : 0.f;
        #pragma unroll
        for (int m = 4; m > 0; m >>= 1) v += __shfl_down(v, m, 32);
        if (threadIdx.x == 0) partials[blockIdx.x] = v;   // deterministic per block
    }
}

// ------------------------- loss + perplexity scalars ------------------------
__global__ __launch_bounds__(1024) void vq_finalize_kernel(
    const unsigned int* __restrict__ counts, const float* __restrict__ partials,
    float* __restrict__ scal)
{
    __shared__ float ws1[32];
    __shared__ float totsq_s;
    int t = threadIdx.x;

    float s = 0.f;
    for (int i = t; i < GATHER_BLOCKS; i += 1024) s += partials[i];
    #pragma unroll
    for (int m = 16; m > 0; m >>= 1) s += __shfl_down(s, m, 32);
    if ((t & 31) == 0) ws1[t >> 5] = s;
    __syncthreads();
    if (t == 0) {
        float v = 0.f;
        for (int i = 0; i < 32; ++i) v += ws1[i];
        totsq_s = v;
    }
    __syncthreads();

    float p    = (float)counts[t] * (1.0f / 65536.0f);
    float term = p * logf(p + 1e-10f);
    #pragma unroll
    for (int m = 16; m > 0; m >>= 1) term += __shfl_down(term, m, 32);
    if ((t & 31) == 0) ws1[t >> 5] = term;
    __syncthreads();
    if (t == 0) {
        float H = 0.f;
        for (int i = 0; i < 32; ++i) H += ws1[i];
        scal[0] = 1.25f * totsq_s * (1.0f / 8388608.0f);  // 1.25 * MSE
        scal[1] = expf(-H);
    }
}

// --------------------------------- launch -----------------------------------
extern "C" void kernel_launch(void* const* d_in, const int* in_sizes, int n_in,
                              void* d_out, int out_size, void* d_ws, size_t ws_size,
                              hipStream_t stream) {
    (void)in_sizes; (void)n_in; (void)out_size; (void)ws_size;
    const float* z  = (const float*)d_in[0];
    const float* cb = (const float*)d_in[1];

    float* out  = (float*)d_out;
    float* qout = out;                     // 8,388,608 elems
    float* idxF = out + Z_ELEMS;           //    65,536 elems
    float* scal = out + Z_ELEMS + N_PTS;   // 2 scalars

    // ws: counts[1024] u32 @0 | cnorm[1024] f32 @4096 | partials[32768] f32 @8192
    //     | cb bf16 hi/lo planes (512 KB) @262144
    unsigned int*   counts   = (unsigned int*)d_ws;
    float*          cnorm    = (float*)((char*)d_ws + 4096);
    float*          partials = (float*)((char*)d_ws + 8192);
    unsigned short* cbs      = (unsigned short*)((char*)d_ws + 262144);

    vq_init_kernel    <<<4,   256, 0, stream>>>(counts);
    vq_cnorm_kernel   <<<4,   256, 0, stream>>>(cb, cnorm);
    vq_cbsplit_kernel <<<512, 256, 0, stream>>>(cb, cbs);
    vq_argmin_kernel  <<<512, 256, 0, stream>>>(z, cbs, cnorm, idxF, counts);
    vq_gather_kernel  <<<GATHER_BLOCKS, 256, 0, stream>>>(z, cb, idxF, qout, partials);
    vq_finalize_kernel<<<1, 1024, 0, stream>>>(counts, partials, scal);
}
